// ProtoNet_83726092469056
// MI455X (gfx1250) — compile-verified, matrix operand order checked by reading, not run on hardware
//
#include <hip/hip_runtime.h>
#include <math.h>

#define EPSV 1e-5f
#define SQRTD 25.298221281347036f

constexpr int DM    = 640;     // feature dim
constexpr int NQR   = 30000;   // query rows
constexpr int NU    = 30025;   // unique rerank rows (query + shot)
constexpr int NPART = 240;     // partial blocks for column sum

typedef __attribute__((ext_vector_type(16))) _Float16 v16h;
typedef __attribute__((ext_vector_type(8)))  float    v8f;

#if __has_builtin(__builtin_amdgcn_global_load_async_to_lds_b128) && \
    __has_builtin(__builtin_amdgcn_s_wait_asynccnt)
#define HAS_ASYNC_LDS 1
typedef int v4i __attribute__((vector_size(16)));
typedef v4i __attribute__((address_space(1))) v4i_g;
typedef v4i __attribute__((address_space(3))) v4i_l;
#else
#define HAS_ASYNC_LDS 0
#endif

// ---------------- reductions ----------------
__device__ __forceinline__ float wred(float v) {
#pragma unroll
  for (int o = 16; o > 0; o >>= 1) v += __shfl_xor(v, o, 32);
  return v;  // all lanes hold result
}

__device__ __forceinline__ float block_sum(float v, volatile float* red) {
  int lane = threadIdx.x & 31, w = threadIdx.x >> 5;
  v = wred(v);
  if (lane == 0) red[w] = v;
  __syncthreads();
  if (threadIdx.x == 0) {
    float t = 0.f;
    int nw = (blockDim.x + 31) >> 5;
    for (int k = 0; k < nw; ++k) t += red[k];
    red[0] = t;
  }
  __syncthreads();
  float t = red[0];
  __syncthreads();
  return t;
}

// ---------------- hyperbolic helpers ----------------
__device__ __forceinline__ float artanhf_(float x) {
  x = fminf(fmaxf(x, -1.f + EPSV), 1.f - EPSV);
  return 0.5f * (log1pf(x) - log1pf(-x));
}

// scalar multiplier s such that e2p(x,c) == s*x, given |x|
__device__ __forceinline__ float e2p_scale(float c, float norm) {
  float sc = sqrtf(c);
  float xn = fmaxf(norm, EPSV);
  float s  = tanhf(sc * xn) / (sc * xn);
  float yn = fmaxf(s * xn, EPSV);
  float mx = 0.999f / sc;
  if (yn > mx) s *= mx / yn;
  return s;
}

// dist_matrix(x,y,c) for one pair given |x|^2, |y|^2, x.y
__device__ __forceinline__ float pdist(float c, float x2, float y2, float dotxy) {
  float sc = sqrtf(c);
  float xy = -dotxy;                       // (-x).y
  float al = 1.f + 2.f * c * xy + c * y2;
  float be = 1.f - c * x2;
  float n2 = al * al * x2 + 2.f * al * be * xy + be * be * y2;
  float dn = 1.f + 2.f * c * xy + c * c * x2 * y2;
  float n  = sqrtf(fmaxf(n2, 0.f)) / fabsf(dn + EPSV);
  return (2.f / sc) * artanhf_(sc * n);
}

// Poincare-mean accumulation of one shared row (len DM)
__device__ void poinc_accum(const float* xrow, float c, float* acc,
                            float* lamsum_s, volatile float* red) {
  int tid = threadIdx.x;
  float x2 = 0.f;
  for (int d = tid; d < DM; d += blockDim.x) x2 += xrow[d] * xrow[d];
  x2 = block_sum(x2, red);
  float s  = e2p_scale(c, sqrtf(fmaxf(x2, 0.f)));
  float y2 = s * s * x2;                   // |e2p(x)|^2
  float kc = 2.f * s / (1.f + c * y2);     // k = kc * x
  float k2 = kc * kc * x2;
  float lam = rsqrtf(fmaxf(1.f - c * k2, 1e-10f));
  for (int d = tid; d < DM; d += blockDim.x) acc[d] += lam * kc * xrow[d];
  if (tid == 0) *lamsum_s += lam;
  __syncthreads();
}

// ---------------- small utility kernels ----------------
__global__ void zero_f(float* p, int n) {
  int i = blockIdx.x * blockDim.x + threadIdx.x;
  if (i < n) p[i] = 0.f;
}

__global__ void __launch_bounds__(256)
colsum_part(const float* __restrict__ query, const float* __restrict__ shot,
            float* __restrict__ partial) {
  int tid = threadIdx.x;
  float a0 = 0.f, a1 = 0.f, a2 = 0.f;
  int c0 = tid, c1 = tid + 256, c2 = tid + 512;
  for (int r = blockIdx.x; r < NU; r += gridDim.x) {
    const float* row = (r < NQR) ? query + (size_t)r * DM
                                 : shot + (size_t)(r - NQR) * DM;
    a0 += row[c0];
    a1 += row[c1];
    if (c2 < DM) a2 += row[c2];
  }
  partial[(size_t)blockIdx.x * DM + c0] = a0;
  partial[(size_t)blockIdx.x * DM + c1] = a1;
  if (c2 < DM) partial[(size_t)blockIdx.x * DM + c2] = a2;
}

__global__ void __launch_bounds__(256)
colsum_red(const float* __restrict__ partial, float* __restrict__ colsum) {
  int c = blockIdx.x * 256 + threadIdx.x;
  if (c >= DM) return;
  float s = 0.f;
  for (int b = 0; b < NPART; ++b) s += partial[(size_t)b * DM + c];
  colsum[c] = s;
}

// ---------------- ctrl MLP -> c_all ----------------
__global__ void __launch_bounds__(256)
ctrl_k(const float* __restrict__ shot, const float* __restrict__ colsum,
       const float* __restrict__ Wc1, const float* __restrict__ bc1,
       const float* __restrict__ Wc2, const float* __restrict__ bc2,
       const float* __restrict__ Wc3, const float* __restrict__ bc3,
       float* __restrict__ meanp, float* __restrict__ c_all) {
  int tid = threadIdx.x;
  __shared__ float s_all[DM];
  __shared__ float s_mp[5 * DM];
  __shared__ float h1[128];
  __shared__ float h2[64];
  for (int t = tid; t < 5 * DM; t += 256) {
    int w = t / DM, d = t - w * DM;
    float s = 0.f;
    for (int s5 = 0; s5 < 5; ++s5) s += shot[(size_t)(s5 * 5 + w) * DM + d];
    s *= 0.2f;
    s_mp[t] = s;
    meanp[t] = s;
  }
  for (int d = tid; d < DM; d += 256) s_all[d] = colsum[d] / (float)NU;
  __syncthreads();
  for (int w = 0; w < 5; ++w) {
    if (tid < 128) {
      float a = 0.f;
      for (int k = 0; k < DM; ++k) {
        float ca = (s_all[k] * 25.f - s_mp[w * DM + k] * 5.f) * (1.f / 20.f);
        a += ca * Wc1[(size_t)k * 128 + tid];
      }
      for (int k = 0; k < DM; ++k)
        a += s_all[k] * Wc1[(size_t)(DM + k) * 128 + tid];
      h1[tid] = fmaxf(a + bc1[tid], 0.f);
    }
    __syncthreads();
    if (tid < 64) {
      float a = 0.f;
      for (int k = 0; k < 128; ++k) a += h1[k] * Wc2[k * 64 + tid];
      h2[tid] = fmaxf(a + bc2[tid], 0.f);
    }
    __syncthreads();
    if (tid == 0) {
      float a = 0.f;
      for (int k = 0; k < 64; ++k) a += h2[k] * Wc3[k];
      c_all[w] = (1.f / (1.f + expf(-(a + bc3[0])))) * 0.1f;  // L/DIV
    }
    __syncthreads();
  }
}

// ---------------- per-class shot prototype ----------------
__global__ void __launch_bounds__(256)
shot_proto(const float* __restrict__ shot, const float* __restrict__ c_all,
           const float* __restrict__ Ws1, const float* __restrict__ bs1,
           const float* __restrict__ Ws2, const float* __restrict__ bs2,
           float* __restrict__ protos, float* __restrict__ protos_x2) {
  int i = blockIdx.x, tid = threadIdx.x;
  __shared__ float G[25], rn2[5], se_s[5], sdm[25], sw[5];
  __shared__ float acc[DM], rowb[DM];
  __shared__ float red[8];
  __shared__ float lamsum_s;
  float c = c_all[i];
  if (tid < 25) {
    int a = tid / 5, b = tid - 5 * a;
    const float* ra = shot + (size_t)(a * 5 + i) * DM;
    const float* rb = shot + (size_t)(b * 5 + i) * DM;
    float d = 0.f;
    for (int k = 0; k < DM; ++k) d += ra[k] * rb[k];
    G[tid] = d;
    if (a == b) rn2[a] = d;
  }
  __syncthreads();
  if (tid < 5) se_s[tid] = e2p_scale(c, sqrtf(fmaxf(rn2[tid], 0.f)));
  __syncthreads();
  if (tid < 25) {
    int a = tid / 5, b = tid - 5 * a;
    float x2 = se_s[a] * se_s[a] * rn2[a];
    float y2 = se_s[b] * se_s[b] * rn2[b];
    float dt = se_s[a] * se_s[b] * G[tid];
    sdm[tid] = pdist(c, x2, y2, dt) / SQRTD;
  }
  __syncthreads();
  if (tid == 0) {
    float sm[25], mx = -3.4e38f, se = 0.f;
    for (int j = 0; j < 25; ++j) mx = fmaxf(mx, -sdm[j]);
    for (int j = 0; j < 25; ++j) { sm[j] = expf(-sdm[j] - mx); se += sm[j]; }
    for (int j = 0; j < 25; ++j) sm[j] /= se;
    float hh[5];
    for (int t = 0; t < 5; ++t) {
      float a = 0.f;
      for (int k = 0; k < 25; ++k) a += sm[k] * Ws1[k * 5 + t];
      hh[t] = fmaxf(a + bs1[t], 0.f);
    }
    float oo[5]; mx = -3.4e38f;
    for (int t = 0; t < 5; ++t) {
      float a = 0.f;
      for (int k = 0; k < 5; ++k) a += hh[k] * Ws2[k * 5 + t];
      oo[t] = a + bs2[t];
      mx = fmaxf(mx, oo[t]);
    }
    se = 0.f;
    for (int t = 0; t < 5; ++t) { oo[t] = expf(oo[t] - mx); se += oo[t]; }
    for (int t = 0; t < 5; ++t) sw[t] = oo[t] / se;
    lamsum_s = 0.f;
  }
  for (int d = tid; d < DM; d += 256) acc[d] = 0.f;
  __syncthreads();
  for (int r = 0; r < 5; ++r) {
    float wsc = sw[r] * 5.f;  // sw * SHOT
    const float* src = shot + (size_t)(r * 5 + i) * DM;
    for (int d = tid; d < DM; d += 256) rowb[d] = src[d] * wsc;
    __syncthreads();
    poinc_accum(rowb, c, acc, &lamsum_s, red);
  }
  float ls = lamsum_s;
  for (int d = tid; d < DM; d += 256) rowb[d] = acc[d] / ls;
  __syncthreads();
  float m2 = 0.f;
  for (int d = tid; d < DM; d += 256) m2 += rowb[d] * rowb[d];
  m2 = block_sum(m2, red);
  float div = 1.f + sqrtf(fmaxf(1.f - c * m2, 1e-10f));
  float x2o = 0.f;
  for (int d = tid; d < DM; d += 256) {
    float t = rowb[d] / div;
    protos[(size_t)i * DM + d] = t;
    x2o += t * t;
  }
  x2o = block_sum(x2o, red);
  if (tid == 0) protos_x2[i] = x2o;
}

// ---------------- WMMA GEMM: C[M,640] = A[M,640] @ B[640,640] + bias ----
// One block owns a 16-wide N slice of B (staged once into LDS, async on
// CDNA5) and 8 consecutive 16-row M tiles (one per wave).
__global__ void __launch_bounds__(256)
wmma_gemm(const float* __restrict__ A0, int M0, const float* __restrict__ A1,
          const float* __restrict__ B, const float* __restrict__ bias,
          float* __restrict__ C, int M, int relu_a) {
  const int NT = DM / 16;
  __shared__ float sB[DM * 16];  // 40 KB: B[:, nt*16 .. nt*16+16)
  int tid = threadIdx.x;
  int wave = tid >> 5, lane = tid & 31;
  int nt = blockIdx.x % NT;
  int mg = blockIdx.x / NT;

  // ---- stage the B column slice into LDS ----
  {
    const float* gB = B + nt * 16;
#if HAS_ASYNC_LDS
    for (int cch = tid; cch < DM * 4; cch += 256) {  // 16-byte chunks
      int k = cch >> 2, o = (cch & 3) * 4;
      __builtin_amdgcn_global_load_async_to_lds_b128(
          (v4i_g*)(gB + (size_t)k * DM + o),
          (v4i_l*)(&sB[k * 16 + o]), 0, 0);
    }
    __builtin_amdgcn_s_wait_asynccnt(0);
#else
    for (int cch = tid; cch < DM * 4; cch += 256) {
      int k = cch >> 2, o = (cch & 3) * 4;
      *(float4*)(&sB[k * 16 + o]) = *(const float4*)(gB + (size_t)k * DM + o);
    }
#endif
  }
  __syncthreads();

  int mt = mg * 8 + wave;
  if (mt * 16 >= M) return;  // after the barrier: safe to exit

  int half = lane >> 4, lq = lane & 15;
  int m = mt * 16 + lq;
  if (m >= M) m = M - 1;  // clamped rows never stored
  const float* arow = (m < M0) ? (A0 + (size_t)m * DM)
                               : (A1 + (size_t)(m - M0) * DM);
  int nb = nt * 16 + lq;
  v8f acc = {};
#pragma unroll 4
  for (int k0 = 0; k0 < DM; k0 += 32) {
    __builtin_prefetch(arow + k0 + 32, 0, 1);
    v16h af, bf;
#pragma unroll
    for (int v = 0; v < 8; ++v) {
      // 16-bit A-fragment K packing per CDNA5 ISA 16x32 layout
      int kk = ((v < 4) ? (2 * v) : (16 + 2 * (v - 4))) + 8 * half;
      float a0 = arow[k0 + kk], a1 = arow[k0 + kk + 1];
      if (relu_a) { a0 = fmaxf(a0, 0.f); a1 = fmaxf(a1, 0.f); }
      af[2 * v]     = (_Float16)a0;
      af[2 * v + 1] = (_Float16)a1;
      bf[2 * v]     = (_Float16)sB[(k0 + kk) * 16 + lq];
      bf[2 * v + 1] = (_Float16)sB[(k0 + kk + 1) * 16 + lq];
    }
    acc = __builtin_amdgcn_wmma_f32_16x16x32_f16(false, af, false, bf,
                                                 (short)0, acc, false, false);
  }
#pragma unroll
  for (int r = 0; r < 8; ++r) {
    int mo = mt * 16 + r + 8 * half;  // C layout: VGPR r -> M = r + 8*half
    if (mo < M) C[(size_t)mo * DM + nb] = acc[r] + bias[nb];
  }
}

// ---------------- fused 5-class support distances ----------------
__global__ void __launch_bounds__(256)
dist_support(const float* __restrict__ qproj, const float* __restrict__ protos,
             const float* __restrict__ protos_x2, const float* __restrict__ c_all,
             float* __restrict__ dis) {
  int wave = threadIdx.x >> 5, lane = threadIdx.x & 31;
  int row = blockIdx.x * 8 + wave;
  if (row >= NU) return;
  const float* y = qproj + (size_t)row * DM;
  float y2 = 0.f, dd[5] = {0.f, 0.f, 0.f, 0.f, 0.f};
  for (int t = lane; t < DM; t += 32) {
    float v = y[t];
    y2 += v * v;
    dd[0] += protos[t] * v;
    dd[1] += protos[DM + t] * v;
    dd[2] += protos[2 * DM + t] * v;
    dd[3] += protos[3 * DM + t] * v;
    dd[4] += protos[4 * DM + t] * v;
  }
  y2 = wred(y2);
#pragma unroll
  for (int i = 0; i < 5; ++i) dd[i] = wred(dd[i]);
  if (lane < 5) {
    float c = c_all[lane];
    float s = e2p_scale(c, sqrtf(fmaxf(y2, 0.f)));
    float d = pdist(c, protos_x2[lane], s * s * y2, s * dd[lane]);
    dis[(size_t)lane * NU + row] = d;
  }
}

// ---------------- per-class top-32 (stable order) + row sums ----------------
__global__ void __launch_bounds__(256)
select_top(const float* __restrict__ dis, int* __restrict__ top_idx,
           float* __restrict__ top_val, float* __restrict__ Ssum,
           float* __restrict__ sum64) {
  int i = blockIdx.x, tid = threadIdx.x;
  const float* row = dis + (size_t)i * NU;
  __shared__ float sv[256];
  __shared__ int   si[256];
  __shared__ float red[8];
  float s = 0.f;
  for (int j = tid; j < NU; j += 256) s += row[j];
  s = block_sum(s, red);
  float s64 = (tid < 64) ? row[tid] : 0.f;
  s64 = block_sum(s64, red);
  if (tid == 0) { Ssum[i] = s; sum64[i] = s64; }
  float pv = -3.4e38f;
  int pi = -1;
  for (int r = 0; r < 32; ++r) {
    float bv = 3.4e38f;
    int bi = NU;
    for (int j = tid; j < NU; j += 256) {
      float v = row[j];
      if (v > pv || (v == pv && j > pi)) {        // strictly after previous pick
        if (v < bv || (v == bv && j < bi)) { bv = v; bi = j; }
      }
    }
    sv[tid] = bv; si[tid] = bi;
    __syncthreads();
    for (int o = 128; o > 0; o >>= 1) {
      if (tid < o) {
        if (sv[tid + o] < sv[tid] ||
            (sv[tid + o] == sv[tid] && si[tid + o] < si[tid])) {
          sv[tid] = sv[tid + o];
          si[tid] = si[tid + o];
        }
      }
      __syncthreads();
    }
    pv = sv[0]; pi = si[0];
    if (tid == 0) { top_idx[i * 32 + r] = pi; top_val[i * 32 + r] = pv; }
    __syncthreads();
  }
}

// ---------------- per-class rerank MLP -> instance weights ----------------
__global__ void __launch_bounds__(256)
rerank_mlp(const float* __restrict__ dis, const int* __restrict__ top_idx,
           const float* __restrict__ top_val, const float* __restrict__ Ssum,
           const float* __restrict__ sum64, const float* __restrict__ Wr1,
           const float* __restrict__ br1, const float* __restrict__ Wr2,
           const float* __restrict__ br2, float* __restrict__ i_w,
           float* __restrict__ onw_out, int* __restrict__ uidx) {
  int i = blockIdx.x, tid = threadIdx.x;
  __shared__ float rin[130], h[64], outv[65], nid[64], nod[64];
  if (tid < 64) {
    int u = tid >> 1;                      // duplicated (MH) columns interleave
    int col = top_idx[i * 32 + u];
    nid[tid] = top_val[i * 32 + u];
    float a = 0.f;
    for (int r = 0; r < 5; ++r)
      if (r != i) a += dis[(size_t)r * NU + col];
    nod[tid] = a * 0.25f;
    uidx[i * 64 + tid] = col;
  }
  __syncthreads();
  if (tid == 0) {
    float mx = -3.4e38f, se = 0.f;
    for (int j = 0; j < 64; ++j) mx = fmaxf(mx, -nid[j] / SQRTD);
    for (int j = 0; j < 64; ++j) { float e = expf(-nid[j] / SQRTD - mx); rin[j] = e; se += e; }
    for (int j = 0; j < 64; ++j) rin[j] /= se;
    mx = -3.4e38f; se = 0.f;
    for (int j = 0; j < 64; ++j) mx = fmaxf(mx, -nod[j] / SQRTD);
    for (int j = 0; j < 64; ++j) { float e = expf(-nod[j] / SQRTD - mx); rin[64 + j] = e; se += e; }
    for (int j = 0; j < 64; ++j) rin[64 + j] /= se;
    float tsum = 0.f;
    for (int u = 0; u < 32; ++u) tsum += top_val[i * 32 + u];
    rin[128] = (2.f * Ssum[i] - 2.f * tsum) / 59986.f;  // o_i_d
    float tot = 0.f;
    for (int r = 0; r < 5; ++r) {
      if (r == i) continue;
      if (r < i) {
        float g = 0.f;
        for (int u = 0; u < 32; ++u) g += dis[(size_t)r * NU + top_idx[i * 32 + u]];
        tot += 2.f * Ssum[r] - 2.f * g;
      } else {
        tot += 2.f * Ssum[r] - sum64[r];   // columns R: of duplicated array
      }
    }
    rin[129] = tot / (4.f * 59986.f);       // o_o_d
  }
  __syncthreads();
  if (tid < 64) {
    float a = 0.f;
    for (int k = 0; k < 130; ++k) a += rin[k] * Wr1[k * 64 + tid];
    h[tid] = fmaxf(a + br1[tid], 0.f);
  }
  __syncthreads();
  if (tid < 65) {
    float a = 0.f;
    for (int k = 0; k < 64; ++k) a += h[k] * Wr2[k * 65 + tid];
    outv[tid] = a + br2[tid];
  }
  __syncthreads();
  if (tid == 0) {
    float mx = -3.4e38f, se = 0.f, e[64];
    for (int j = 0; j < 64; ++j) mx = fmaxf(mx, outv[j]);
    for (int j = 0; j < 64; ++j) { e[j] = expf(outv[j] - mx); se += e[j]; }
    float on = 1.f / (1.f + expf(-outv[64]));
    onw_out[i] = on;
    for (int j = 0; j < 64; ++j) i_w[i * 64 + j] = (e[j] / se) * on * 65.f;
  }
}

// ---------------- per-class test prototype (v_proj on the fly) ----------------
__global__ void __launch_bounds__(256)
build_proto(const float* __restrict__ qproj, const int* __restrict__ uidx,
            const float* __restrict__ i_w, const float* __restrict__ onw,
            const float* __restrict__ Wv, const float* __restrict__ bv,
            const float* __restrict__ Wfc, const float* __restrict__ bfc,
            const float* __restrict__ ln_g, const float* __restrict__ ln_b,
            const float* __restrict__ meanp, const float* __restrict__ c_all,
            float* __restrict__ tproto, float* __restrict__ tpx2) {
  int i = blockIdx.x, tid = threadIdx.x;
  __shared__ float xrow[DM], vrow[DM], yrow[DM], acc[DM];
  __shared__ float red[8];
  __shared__ float lamsum_s;
  float c = c_all[i];
  for (int d = tid; d < DM; d += 256) acc[d] = 0.f;
  if (tid == 0) lamsum_s = 0.f;
  __syncthreads();
  for (int j = 0; j < 64; ++j) {
    const float* q = qproj + (size_t)uidx[i * 64 + j] * DM;
    for (int d = tid; d < DM; d += 256) xrow[d] = fmaxf(q[d], 0.f);
    __syncthreads();
    float w = i_w[i * 64 + j];
    for (int d = tid; d < DM; d += 256) {
      float a = 0.f;
      for (int k = 0; k < DM; ++k) a += xrow[k] * Wv[(size_t)k * DM + d];
      vrow[d] = (a + bv[d]) * w;            // v_proj row * instance weight
    }
    __syncthreads();
    for (int d = tid; d < DM; d += 256) {
      float a = 0.f;
      for (int k = 0; k < DM; ++k) a += vrow[k] * Wfc[(size_t)k * DM + d];
      yrow[d] = a + bfc[d];
    }
    __syncthreads();
    float m = 0.f;
    for (int d = tid; d < DM; d += 256) m += yrow[d];
    m = block_sum(m, red) / (float)DM;
    float var = 0.f;
    for (int d = tid; d < DM; d += 256) { float t = yrow[d] - m; var += t * t; }
    var = block_sum(var, red) / (float)DM;
    float inv = rsqrtf(var + 1e-5f);
    for (int d = tid; d < DM; d += 256)
      yrow[d] = (yrow[d] - m) * inv * ln_g[d] + ln_b[d];
    __syncthreads();
    poinc_accum(yrow, c, acc, &lamsum_s, red);
  }
  // mean_i = mean_proto[i] * (1 - onw) * (R + 1)
  float msc = (1.f - onw[i]) * 65.f;
  for (int d = tid; d < DM; d += 256) yrow[d] = meanp[(size_t)i * DM + d] * msc;
  __syncthreads();
  poinc_accum(yrow, c, acc, &lamsum_s, red);
  float ls = lamsum_s;
  for (int d = tid; d < DM; d += 256) yrow[d] = acc[d] / ls;
  __syncthreads();
  float m2 = 0.f;
  for (int d = tid; d < DM; d += 256) m2 += yrow[d] * yrow[d];
  m2 = block_sum(m2, red);
  float div = 1.f + sqrtf(fmaxf(1.f - c * m2, 1e-10f));
  float x2o = 0.f;
  for (int d = tid; d < DM; d += 256) {
    float t = yrow[d] / div;
    tproto[(size_t)i * DM + d] = t;
    x2o += t * t;
  }
  x2o = block_sum(x2o, red);
  if (tid == 0) tpx2[i] = x2o;
}

// ---------------- final query distances ----------------
__global__ void __launch_bounds__(256)
final_dist(const float* __restrict__ query, const float* __restrict__ tproto,
           const float* __restrict__ tpx2, const float* __restrict__ c_all,
           float* __restrict__ out) {
  int wave = threadIdx.x >> 5, lane = threadIdx.x & 31;
  int row = blockIdx.x * 8 + wave;
  if (row >= NQR) return;
  const float* y = query + (size_t)row * DM;
  float y2 = 0.f, dd[5] = {0.f, 0.f, 0.f, 0.f, 0.f};
  for (int t = lane; t < DM; t += 32) {
    float v = y[t];
    y2 += v * v;
    dd[0] += tproto[t] * v;
    dd[1] += tproto[DM + t] * v;
    dd[2] += tproto[2 * DM + t] * v;
    dd[3] += tproto[3 * DM + t] * v;
    dd[4] += tproto[4 * DM + t] * v;
  }
  y2 = wred(y2);
#pragma unroll
  for (int k = 0; k < 5; ++k) dd[k] = wred(dd[k]);
  if (lane < 5) {
    float c = c_all[lane];
    float s = e2p_scale(c, sqrtf(fmaxf(y2, 0.f)));
    float d = pdist(c, tpx2[lane], s * s * y2, s * dd[lane]);
    out[(size_t)row * 5 + lane] = -d / 64.f;  // -dist / TEMP
  }
}

// ---------------- host orchestration ----------------
extern "C" void kernel_launch(void* const* d_in, const int* in_sizes, int n_in,
                              void* d_out, int out_size, void* d_ws,
                              size_t ws_size, hipStream_t stream) {
  (void)in_sizes; (void)n_in; (void)out_size; (void)ws_size;
  const float* shot  = (const float*)d_in[0];
  const float* query = (const float*)d_in[1];
  const float* Wq  = (const float*)d_in[4];
  const float* bq  = (const float*)d_in[5];
  const float* Wv  = (const float*)d_in[6];
  const float* bv  = (const float*)d_in[7];
  const float* Wfc = (const float*)d_in[8];
  const float* bfc = (const float*)d_in[9];
  const float* lng = (const float*)d_in[10];
  const float* lnb = (const float*)d_in[11];
  const float* Wc1 = (const float*)d_in[12];
  const float* bc1 = (const float*)d_in[13];
  const float* Wc2 = (const float*)d_in[14];
  const float* bc2 = (const float*)d_in[15];
  const float* Wc3 = (const float*)d_in[16];
  const float* bc3 = (const float*)d_in[17];
  const float* Ws1 = (const float*)d_in[18];
  const float* bs1 = (const float*)d_in[19];
  const float* Ws2 = (const float*)d_in[20];
  const float* bs2 = (const float*)d_in[21];
  const float* Wr1 = (const float*)d_in[22];
  const float* br1 = (const float*)d_in[23];
  const float* Wr2 = (const float*)d_in[24];
  const float* br2 = (const float*)d_in[25];

  float* w = (float*)d_ws;
  size_t off = 0;
  float* qproj   = w + off; off += (size_t)NU * DM;
  float* dis     = w + off; off += (size_t)5 * NU;
  float* partial = w + off; off += (size_t)NPART * DM;
  float* colsum  = w + off; off += DM;
  float* meanp   = w + off; off += 5 * DM;
  float* c_all   = w + off; off += 8;
  float* protos  = w + off; off += 5 * DM;
  float* protx2  = w + off; off += 8;
  float* topval  = w + off; off += 5 * 32;
  float* Ssum    = w + off; off += 8;
  float* sum64   = w + off; off += 8;
  float* iw      = w + off; off += 5 * 64;
  float* onw     = w + off; off += 8;
  float* tproto  = w + off; off += 5 * DM;
  float* tpx2    = w + off; off += 8;
  int* topidx = (int*)(w + off); off += 5 * 32;
  int* uidx   = (int*)(w + off); off += 5 * 64;

  zero_f<<<3, 256, 0, stream>>>(colsum, DM);
  colsum_part<<<NPART, 256, 0, stream>>>(query, shot, partial);
  colsum_red<<<3, 256, 0, stream>>>(partial, colsum);
  ctrl_k<<<1, 256, 0, stream>>>(shot, colsum, Wc1, bc1, Wc2, bc2, Wc3, bc3,
                                meanp, c_all);
  shot_proto<<<5, 256, 0, stream>>>(shot, c_all, Ws1, bs1, Ws2, bs2, protos,
                                    protx2);
  int mgroups = (NU + 127) / 128;          // 8 waves x 16 rows per block
  int nblocks = mgroups * (DM / 16);
  wmma_gemm<<<nblocks, 256, 0, stream>>>(query, NQR, shot, Wq, bq, qproj, NU, 0);
  dist_support<<<(NU + 7) / 8, 256, 0, stream>>>(qproj, protos, protx2, c_all,
                                                 dis);
  select_top<<<5, 256, 0, stream>>>(dis, topidx, topval, Ssum, sum64);
  rerank_mlp<<<5, 256, 0, stream>>>(dis, topidx, topval, Ssum, sum64, Wr1, br1,
                                    Wr2, br2, iw, onw, uidx);
  build_proto<<<5, 256, 0, stream>>>(qproj, uidx, iw, onw, Wv, bv, Wfc, bfc,
                                     lng, lnb, meanp, c_all, tproto, tpx2);
  final_dist<<<(NQR + 7) / 8, 256, 0, stream>>>(query, tproto, tpx2, c_all,
                                                (float*)d_out);
}
